// CKAFormer_85487029059851
// MI455X (gfx1250) — matrix operand order repaired and checked by reading, not compiled, not run on hardware
//
#include <hip/hip_runtime.h>
#include <hip/hip_bf16.h>
#include <math.h>

#define DEPTH   6
#define GAMMA   1e-4f
#define DIM     1024
#define HID     16
#define ODIM    64
#define NROWS   16384
#define KSPLIT  16
#define TSTRIDE 40   // halves; 80 bytes/row -> 16B aligned rows, conflict-skewed

typedef unsigned short u16t;
typedef unsigned int   u32t;
typedef unsigned long long u64t;
typedef __attribute__((ext_vector_type(16))) __bf16 v16bf;
typedef __attribute__((ext_vector_type(8)))  __bf16 v8bf;
typedef __attribute__((ext_vector_type(8)))  float  v8f;

#define WMMA_BF16(A, B, C) \
    __builtin_amdgcn_wmma_f32_16x16x32_bf16(false, (A), false, (B), (short)0, (C), false, false)

// Native RNE float->bf16 (backend picks the hw cvt op).
__device__ __forceinline__ u16t f2bf(float f) {
    __bf16 h = (__bf16)f;
    return __builtin_bit_cast(unsigned short, h);
}

// ---------------------------------------------------------------------------
// Tensor Data Mover staging (natural row-major tile, 128 rows x 32 halves),
// with TDM padding reproducing the TSTRIDE=40 skew (16 dwords data + 4 pad).
// This toolchain exposes the 6-arg builtin:
//   (uint32x4 g0, int32x8 g1, int32x4 g2, int32x4 g3, int32x8 g4, i32 cpol)
// ---------------------------------------------------------------------------
#if __has_builtin(__builtin_amdgcn_tensor_load_to_lds) && \
    __has_builtin(__builtin_amdgcn_s_wait_tensorcnt)
#define HAVE_TDM 1
typedef __attribute__((ext_vector_type(4))) unsigned int v4u;
typedef __attribute__((ext_vector_type(8))) int v8i;
typedef __attribute__((ext_vector_type(4))) int v4i;

__device__ __forceinline__ void tdm_stageN(u16t* T, const u16t* g, int gStride,
                                           int r0, int ks, u32t totalRows) {
    u64t ga  = (u64t)(g + (size_t)r0 * gStride + ks);
    u32t lds = (u32t)(u64t)T;            // generic LDS addr low bits = LDS offset
    v4u g0;
    g0[0] = 1u;                                            // count=1 (valid D#)
    g0[1] = lds;                                           // lds_addr
    g0[2] = (u32t)ga;                                      // global_addr[31:0]
    g0[3] = (u32t)((ga >> 32) & 0x01ffffffu) | (2u << 30); // addr[56:32] | type=2
    v8i g1;
    g1[0] = (int)((1u << 16) | (1u << 20) | (3u << 22) | (3u << 25));
            // data_size=2B | pad_enable | pad_interval=16dw | pad_amount=4dw
    g1[1] = (int)(((u32t)gStride & 0xffffu) << 16);        // tensor_dim0 lo16
    g1[2] = (int)((((u32t)gStride >> 16) & 0xffffu) |
                  ((totalRows & 0xffffu) << 16));          // dim0 hi | dim1 lo
    g1[3] = (int)(((totalRows >> 16) & 0xffffu) | (32u << 16)); // dim1 hi | tile_dim0=32
    g1[4] = 128;                                           // tile_dim1=128 rows
    g1[5] = gStride;                                       // tensor_dim0_stride
    g1[6] = 0;
    g1[7] = 0;
    v4i z4 = {0, 0, 0, 0};
    v8i z8 = {0, 0, 0, 0, 0, 0, 0, 0};
    __builtin_amdgcn_tensor_load_to_lds(g0, g1, z4, z4, z8, 0);
}
#endif

// ---------------------------------------------------------------------------
// LDS staging helpers
// ---------------------------------------------------------------------------

// Transposed stage, register-packed: T[cc*TSTRIDE+kk] = g[(k0+kk)*gS + c0+cc].
// Each thread handles a 2(k) x 8(col) sub-block -> 8x ds_store_b32.
template <int NCOLS>
__device__ __forceinline__ void stageT2(u16t* T, const u16t* __restrict__ g,
                                        int gStride, int k0, int c0, int t) {
    constexpr int CB = NCOLS / 8;       // 8-col blocks per k-row
    constexpr int NT = 16 * CB;         // threads used (16 k-pairs)
    if (NT < 256 && t >= NT) return;
    int kp = t / CB;                    // k-pair 0..15
    int c8 = (t % CB) * 8;
    const u16t* src = g + (size_t)(k0 + 2 * kp) * gStride + c0 + c8;
    uint4 q0 = *reinterpret_cast<const uint4*>(src);
    uint4 q1 = *reinterpret_cast<const uint4*>(src + gStride);
    __builtin_prefetch(src + 32 * (size_t)gStride, 0, 3);   // next k-step
    u32t a[4] = {q0.x, q0.y, q0.z, q0.w};
    u32t b[4] = {q1.x, q1.y, q1.z, q1.w};
#pragma unroll
    for (int e = 0; e < 8; ++e) {
        u32t h0 = (a[e >> 1] >> ((e & 1) * 16)) & 0xffffu;
        u32t h1 = (b[e >> 1] >> ((e & 1) * 16)) & 0xffffu;
        *reinterpret_cast<u32t*>(T + (c8 + e) * TSTRIDE + 2 * kp) = h0 | (h1 << 16);
    }
}

// Natural stage (fallback when TDM builtin unavailable).
__device__ __forceinline__ void stageN(u16t* T, const u16t* __restrict__ g,
                                       int gStride, int r0, int ks, int t) {
#pragma unroll
    for (int i = 0; i < 2; ++i) {
        int q   = t + 256 * i;          // 0..511
        int row = q >> 2;
        int k8  = (q & 3) * 8;
        uint4 d = *reinterpret_cast<const uint4*>(g + (size_t)(r0 + row) * gStride + ks + k8);
        *reinterpret_cast<uint4*>(T + row * TSTRIDE + k8) = d;
    }
}

// A fragment (16x32 bf16, MxK): lane<16 -> K 0..7 & 16..23; lane>=16 -> K 8..15 & 24..31
__device__ __forceinline__ v16bf loadA(const u16t* T, int mt, int lane) {
    int row = mt * 16 + (lane & 15);
    int klo = (lane >> 4) * 8;
    v8bf lo = *reinterpret_cast<const v8bf*>(T + row * TSTRIDE + klo);
    v8bf hi = *reinterpret_cast<const v8bf*>(T + row * TSTRIDE + klo + 16);
    return __builtin_shufflevector(lo, hi, 0,1,2,3,4,5,6,7,8,9,10,11,12,13,14,15);
}

// B fragment (32x16 bf16, KxN): lane<16 -> K 0..7 & 16..23; lane>=16 -> K 8..15 & 24..31; N=lane&15
__device__ __forceinline__ v16bf loadB(const u16t* T, int nt, int lane) {
    int col = nt * 16 + (lane & 15);
    int kb  = (lane >> 4) * 8;
    v8bf lo = *reinterpret_cast<const v8bf*>(T + col * TSTRIDE + kb);
    v8bf hi = *reinterpret_cast<const v8bf*>(T + col * TSTRIDE + kb + 16);
    return __builtin_shufflevector(lo, hi, 0,1,2,3,4,5,6,7,8,9,10,11,12,13,14,15);
}

// ---------------------------------------------------------------------------
// Kernel 1: row L2-norm -> inv scale + bf16 normalized X
// ---------------------------------------------------------------------------
__global__ __launch_bounds__(256) void k_norm(const float* __restrict__ X,
                                              float* __restrict__ invn,
                                              u16t* __restrict__ Xbf) {
    __shared__ float red[256];
    int r = blockIdx.x, t = threadIdx.x;
    float4 x4 = *reinterpret_cast<const float4*>(X + (size_t)r * DIM + t * 4);
    red[t] = x4.x * x4.x + x4.y * x4.y + x4.z * x4.z + x4.w * x4.w;
    __syncthreads();
    for (int off = 128; off > 0; off >>= 1) {
        if (t < off) red[t] += red[t + off];
        __syncthreads();
    }
    float inv = rsqrtf(red[0]);
    if (t == 0) invn[r] = inv;
    u32t p0 = (u32t)f2bf(x4.x * inv) | ((u32t)f2bf(x4.y * inv) << 16);
    u32t p1 = (u32t)f2bf(x4.z * inv) | ((u32t)f2bf(x4.w * inv) << 16);
    uint2 pk; pk.x = p0; pk.y = p1;
    *reinterpret_cast<uint2*>(Xbf + (size_t)r * DIM + t * 4) = pk;
}

// ---------------------------------------------------------------------------
// Kernel 2: MLP (+ optional softmax). Wave per row (wave32).
// ---------------------------------------------------------------------------
__global__ __launch_bounds__(256) void k_mlp(const float* __restrict__ X,
                                             const float* __restrict__ invn,
                                             const float* __restrict__ W1,
                                             const float* __restrict__ b1,
                                             const float* __restrict__ W2,
                                             const float* __restrict__ b2,
                                             u16t* __restrict__ Pbf,
                                             float* __restrict__ logitsOut,
                                             int doSoftmax) {
    int wave = threadIdx.x >> 5, lane = threadIdx.x & 31;
    int r = blockIdx.x * 8 + wave;
    float inv = invn ? invn[r] : 1.0f;
    const float* xrow = X + (size_t)r * DIM;
    float acc[HID];
#pragma unroll
    for (int h = 0; h < HID; ++h) acc[h] = 0.f;
    for (int j = 0; j < DIM / 32; ++j) {
        int k   = j * 32 + lane;
        float xs = xrow[k] * inv;
        const float4* w4 = reinterpret_cast<const float4*>(W1 + k * HID);
        float4 w[4]; w[0] = w4[0]; w[1] = w4[1]; w[2] = w4[2]; w[3] = w4[3];
        const float* wf = reinterpret_cast<const float*>(w);
#pragma unroll
        for (int h = 0; h < HID; ++h) acc[h] += xs * wf[h];
    }
#pragma unroll
    for (int h = 0; h < HID; ++h) {
        float v = acc[h];
        v += __shfl_xor(v, 16, 32);
        v += __shfl_xor(v, 8, 32);
        v += __shfl_xor(v, 4, 32);
        v += __shfl_xor(v, 2, 32);
        v += __shfl_xor(v, 1, 32);
        acc[h] = fmaxf(v + b1[h], 0.f);   // hidden activation, all lanes
    }
    int o0 = lane, o1 = lane + 32;
    float l0 = b2[o0], l1 = b2[o1];
#pragma unroll
    for (int h = 0; h < HID; ++h) {
        l0 += acc[h] * W2[h * ODIM + o0];
        l1 += acc[h] * W2[h * ODIM + o1];
    }
    if (doSoftmax) {
        float m = fmaxf(l0, l1);
        m = fmaxf(m, __shfl_xor(m, 16, 32));
        m = fmaxf(m, __shfl_xor(m, 8, 32));
        m = fmaxf(m, __shfl_xor(m, 4, 32));
        m = fmaxf(m, __shfl_xor(m, 2, 32));
        m = fmaxf(m, __shfl_xor(m, 1, 32));
        float e0 = __expf(l0 - m), e1 = __expf(l1 - m);
        float s = e0 + e1;
        s += __shfl_xor(s, 16, 32);
        s += __shfl_xor(s, 8, 32);
        s += __shfl_xor(s, 4, 32);
        s += __shfl_xor(s, 2, 32);
        s += __shfl_xor(s, 1, 32);
        float is = 1.0f / s;
        Pbf[(size_t)r * ODIM + o0] = f2bf(e0 * is);
        Pbf[(size_t)r * ODIM + o1] = f2bf(e1 * is);
    } else {
        logitsOut[(size_t)r * ODIM + o0] = l0;
        logitsOut[(size_t)r * ODIM + o1] = l1;
    }
}

// ---------------------------------------------------------------------------
// Kernel 3: Gn = -(Xn^T @ Xn) in bf16. 128x128 tile/block, K=16384, WMMA bf16.
// ---------------------------------------------------------------------------
__global__ __launch_bounds__(256) void k_gram(const u16t* __restrict__ Xbf,
                                              u16t* __restrict__ Gnbf) {
    __shared__ __align__(16) u16t Ta[128 * TSTRIDE];
    __shared__ __align__(16) u16t Tb[128 * TSTRIDE];
    int t = threadIdx.x, wave = t >> 5, lane = t & 31;
    int wr = wave >> 1, wc = wave & 1;
    int i0 = blockIdx.y * 128, j0 = blockIdx.x * 128;
    v8f acc[2][4];
#pragma unroll
    for (int mi = 0; mi < 2; ++mi)
#pragma unroll
        for (int ni = 0; ni < 4; ++ni)
#pragma unroll
            for (int e = 0; e < 8; ++e) acc[mi][ni][e] = 0.f;

    for (int k0 = 0; k0 < NROWS; k0 += 32) {
        stageT2<128>(Ta, Xbf, DIM, k0, i0, t);  // A = Xn^T tile (col-of-Xn major)
        stageT2<128>(Tb, Xbf, DIM, k0, j0, t);  // B = Xn tile  (N-major)
        __syncthreads();
        v16bf a0 = loadA(Ta, wr * 2,     lane);
        v16bf a1 = loadA(Ta, wr * 2 + 1, lane);
#pragma unroll
        for (int ni = 0; ni < 4; ++ni) {
            v16bf b = loadB(Tb, wc * 4 + ni, lane);
            acc[0][ni] = WMMA_BF16(a0, b, acc[0][ni]);
            acc[1][ni] = WMMA_BF16(a1, b, acc[1][ni]);
        }
        __syncthreads();
    }
#pragma unroll
    for (int mi = 0; mi < 2; ++mi)
#pragma unroll
        for (int ni = 0; ni < 4; ++ni) {
            int row0 = i0 + (wr * 2 + mi) * 16 + ((lane >> 4) * 8);
            int col  = j0 + wc * 64 + ni * 16 + (lane & 15);
#pragma unroll
            for (int v = 0; v < 8; ++v)
                Gnbf[(size_t)(row0 + v) * DIM + col] = f2bf(-acc[mi][ni][v]);
        }
}

// ---------------------------------------------------------------------------
// Kernel 4: split-K partials of PtX = P^T @ Xn. M=64, N=1024, K split by 16.
// ---------------------------------------------------------------------------
__global__ __launch_bounds__(256) void k_ptx(const u16t* __restrict__ Pbf,
                                             const u16t* __restrict__ Xbf,
                                             float* __restrict__ part) {
    __shared__ __align__(16) u16t Ta[64 * TSTRIDE];
    __shared__ __align__(16) u16t Tb[128 * TSTRIDE];
    int t = threadIdx.x, wave = t >> 5, lane = t & 31;
    int c0 = blockIdx.x * 128;
    int kz = blockIdx.y;
    int kbase = kz * (NROWS / KSPLIT);
    v8f acc[4];
#pragma unroll
    for (int mi = 0; mi < 4; ++mi)
#pragma unroll
        for (int e = 0; e < 8; ++e) acc[mi][e] = 0.f;

    for (int ks = 0; ks < NROWS / KSPLIT; ks += 32) {
        int k0 = kbase + ks;
        stageT2<64>(Ta, Pbf, ODIM, k0, 0, t);   // A = P^T tile
        stageT2<128>(Tb, Xbf, DIM, k0, c0, t);  // B = Xn tile
        __syncthreads();
        v16bf b = loadB(Tb, wave, lane);        // wave = n-tile
#pragma unroll
        for (int mi = 0; mi < 4; ++mi) {
            v16bf a = loadA(Ta, mi, lane);
            acc[mi] = WMMA_BF16(a, b, acc[mi]);
        }
        __syncthreads();
    }
    float* dst = part + (size_t)kz * (ODIM * DIM);
    int col = c0 + wave * 16 + (lane & 15);
#pragma unroll
    for (int mi = 0; mi < 4; ++mi) {
        int row0 = mi * 16 + ((lane >> 4) * 8);
#pragma unroll
        for (int v = 0; v < 8; ++v)
            dst[(size_t)(row0 + v) * DIM + col] = acc[mi][v];
    }
}

__global__ __launch_bounds__(256) void k_ptx_reduce(const float* __restrict__ part,
                                                    u16t* __restrict__ PtXbf) {
    int idx = blockIdx.x * 256 + threadIdx.x;   // 64*1024 total
    float s = 0.f;
#pragma unroll
    for (int z = 0; z < KSPLIT; ++z) s += part[(size_t)z * (ODIM * DIM) + idx];
    PtXbf[idx] = f2bf(s);
}

// ---------------------------------------------------------------------------
// Kernel 5: X_next = Xn + GAMMA*(P@PtX + Xn@(-G)). Fused two K-loops, WMMA bf16.
// A-tiles staged by the Tensor Data Mover (TDM pad = TSTRIDE skew) when available.
// ---------------------------------------------------------------------------
__global__ __launch_bounds__(256) void k_update(const float* __restrict__ Xin,
                                                const float* __restrict__ invn,
                                                const u16t* __restrict__ Xbf,
                                                const u16t* __restrict__ Pbf,
                                                const u16t* __restrict__ PtXbf,
                                                const u16t* __restrict__ Gnbf,
                                                float* __restrict__ Xout) {
    __shared__ __align__(16) u16t Ta[128 * TSTRIDE];
    __shared__ __align__(16) u16t Tb[128 * TSTRIDE];
    int t = threadIdx.x, wave = t >> 5, lane = t & 31;
    int wr = wave >> 1, wc = wave & 1;
    int r0 = blockIdx.y * 128, c0 = blockIdx.x * 128;
    v8f acc[2][4];
#pragma unroll
    for (int mi = 0; mi < 2; ++mi)
#pragma unroll
        for (int ni = 0; ni < 4; ++ni)
#pragma unroll
            for (int e = 0; e < 8; ++e) acc[mi][ni][e] = 0.f;

    // Phase 1: P @ PtX  (K = 64)
    for (int ks = 0; ks < ODIM; ks += 32) {
#ifdef HAVE_TDM
        if (wave == 0) tdm_stageN(Ta, Pbf, ODIM, r0, ks, NROWS);
        stageT2<128>(Tb, PtXbf, DIM, ks, c0, t);
        if (wave == 0) __builtin_amdgcn_s_wait_tensorcnt(0);
#else
        stageN(Ta, Pbf, ODIM, r0, ks, t);
        stageT2<128>(Tb, PtXbf, DIM, ks, c0, t);
#endif
        __syncthreads();
        v16bf a0 = loadA(Ta, wr * 2,     lane);
        v16bf a1 = loadA(Ta, wr * 2 + 1, lane);
#pragma unroll
        for (int ni = 0; ni < 4; ++ni) {
            v16bf b = loadB(Tb, wc * 4 + ni, lane);
            acc[0][ni] = WMMA_BF16(a0, b, acc[0][ni]);
            acc[1][ni] = WMMA_BF16(a1, b, acc[1][ni]);
        }
        __syncthreads();
    }
    // Phase 2: Xn @ (-G)  (K = 1024), G pre-negated so plain accumulate works
    for (int ks = 0; ks < DIM; ks += 32) {
#ifdef HAVE_TDM
        if (wave == 0) tdm_stageN(Ta, Xbf, DIM, r0, ks, NROWS);
        stageT2<128>(Tb, Gnbf, DIM, ks, c0, t);
        if (wave == 0) __builtin_amdgcn_s_wait_tensorcnt(0);
#else
        stageN(Ta, Xbf, DIM, r0, ks, t);
        stageT2<128>(Tb, Gnbf, DIM, ks, c0, t);
#endif
        __syncthreads();
        v16bf a0 = loadA(Ta, wr * 2,     lane);
        v16bf a1 = loadA(Ta, wr * 2 + 1, lane);
#pragma unroll
        for (int ni = 0; ni < 4; ++ni) {
            v16bf b = loadB(Tb, wc * 4 + ni, lane);
            acc[0][ni] = WMMA_BF16(a0, b, acc[0][ni]);
            acc[1][ni] = WMMA_BF16(a1, b, acc[1][ni]);
        }
        __syncthreads();
    }
    // Epilogue: residual in fp32 (Xn recomputed as X*inv). In-place safe.
#pragma unroll
    for (int mi = 0; mi < 2; ++mi)
#pragma unroll
        for (int ni = 0; ni < 4; ++ni) {
            int row0 = r0 + (wr * 2 + mi) * 16 + ((lane >> 4) * 8);
            int col  = c0 + wc * 64 + ni * 16 + (lane & 15);
#pragma unroll
            for (int v = 0; v < 8; ++v) {
                int row = row0 + v;
                float xn = Xin[(size_t)row * DIM + col] * invn[row];
                Xout[(size_t)row * DIM + col] = xn + GAMMA * acc[mi][ni][v];
            }
        }
}

// ---------------------------------------------------------------------------
extern "C" void kernel_launch(void* const* d_in, const int* in_sizes, int n_in,
                              void* d_out, int out_size, void* d_ws, size_t ws_size,
                              hipStream_t stream) {
    (void)in_sizes; (void)n_in; (void)out_size; (void)ws_size;
    const float* X0 = (const float*)d_in[0];
    const float* W1 = (const float*)d_in[1];
    const float* b1 = (const float*)d_in[2];
    const float* W2 = (const float*)d_in[3];
    const float* b2 = (const float*)d_in[4];

    char* w = (char*)d_ws;
    size_t off = 0;
    float* X_a   = (float*)(w + off); off += (size_t)NROWS * DIM * 4;     // 64 MB
    u16t*  Xn_bf = (u16t*)(w + off);  off += (size_t)NROWS * DIM * 2;     // 32 MB
    u16t*  P_bf  = (u16t*)(w + off);  off += (size_t)NROWS * ODIM * 2;    // 2 MB
    u16t*  Gn_bf = (u16t*)(w + off);  off += (size_t)DIM * DIM * 2;       // 2 MB
    float* Ppart = (float*)(w + off); off += (size_t)KSPLIT * ODIM * DIM * 4; // 4 MB
    u16t*  PtXbf = (u16t*)(w + off);  off += (size_t)ODIM * DIM * 2;      // 128 KB
    float* invn  = (float*)(w + off); off += (size_t)NROWS * 4;           // 64 KB

    for (int it = 0; it < DEPTH; ++it) {
        const float* xin = (it == 0) ? X0 : X_a;
        k_norm<<<NROWS, 256, 0, stream>>>(xin, invn, Xn_bf);
        k_mlp<<<NROWS / 8, 256, 0, stream>>>(xin, invn, W1, b1, W2, b2,
                                             P_bf, nullptr, 1);
        k_gram<<<dim3(DIM / 128, DIM / 128), 256, 0, stream>>>(Xn_bf, Gn_bf);
        k_ptx<<<dim3(DIM / 128, KSPLIT), 256, 0, stream>>>(P_bf, Xn_bf, Ppart);
        k_ptx_reduce<<<(ODIM * DIM) / 256, 256, 0, stream>>>(Ppart, PtXbf);
        k_update<<<dim3(DIM / 128, NROWS / 128), 256, 0, stream>>>(
            xin, invn, Xn_bf, P_bf, PtXbf, Gn_bf, X_a);
    }
    // Final head on un-normalized X: logits fp32 -> d_out
    k_mlp<<<NROWS / 8, 256, 0, stream>>>(X_a, nullptr, W1, b1, W2, b2,
                                         nullptr, (float*)d_out, 0);
}
